// MultiQueryAttentionPEZ_21062519619680
// MI455X (gfx1250) — compile-verified, matrix operand order checked by reading, not run on hardware
//
#include <hip/hip_runtime.h>
#include <hip/hip_bf16.h>

typedef __attribute__((ext_vector_type(16))) __bf16 v16bf;
typedef __attribute__((ext_vector_type(8)))  float  v8f;

#define WMMA_BF16(a,b,c) \
  __builtin_amdgcn_wmma_f32_16x16x32_bf16(false,(a),false,(b),(short)0,(c),false,false)

// Problem constants (match reference)
constexpr int B_  = 2;
constexpr int T_  = 2048;
constexpr int C_  = 1024;
constexpr int H_  = 16;
constexpr int HD_ = 64;
constexpr int O_  = C_ + 2 * HD_;   // 1152
constexpr int M_  = B_ * T_;        // 4096 tokens

// ---------------- fp32 -> bf16 conversion ----------------
__global__ void __launch_bounds__(256) cvt_f32_bf16(const float* __restrict__ in,
                                                    __bf16* __restrict__ out, int n) {
  int i = blockIdx.x * blockDim.x + threadIdx.x;
  if (i < n) out[i] = (__bf16)in[i];
}

// ---------------- bf16 GEMM: C[M,N] = A[M,K] * W[N,K]^T, fp32 out ----------
// One wave computes a 32x64 tile: 2 M-subtiles x 4 N-subtiles, K-steps of 32.
__global__ void __launch_bounds__(256) gemm_bf16_f32(const __bf16* __restrict__ A,
                                                     const __bf16* __restrict__ W,
                                                     float* __restrict__ Cout,
                                                     int M, int N, int K) {
  int wave = (blockIdx.x * blockDim.x + threadIdx.x) >> 5;
  int lane = threadIdx.x & 31;
  int nGroups = N >> 6;
  int mGroup = wave / nGroups;
  int nGroup = wave % nGroups;
  if (mGroup * 32 >= M) return;
  int m0 = mGroup * 32, n0 = nGroup * 64;
  int lq  = lane & 15;
  int hs  = (lane >> 4) << 4;   // 0 or 16 : K half selected by lane group
  int hi8 = (lane >> 4) << 3;   // 0 or 8  : row offset in C/D layout

  v8f acc[2][4] = {};

  const __bf16* Ar0 = A + (size_t)(m0 + lq) * K + hs;
  const __bf16* Ar1 = A + (size_t)(m0 + 16 + lq) * K + hs;
  const __bf16* Wr  = W + (size_t)(n0 + lq) * K + hs;

  for (int kk = 0; kk < K; kk += 32) {
    __builtin_prefetch(Ar0 + kk + 256, 0, 1);   // global_prefetch_b8
    v16bf a0 = *(const v16bf*)(Ar0 + kk);
    v16bf a1 = *(const v16bf*)(Ar1 + kk);
    v16bf b0 = *(const v16bf*)(Wr + kk);
    v16bf b1 = *(const v16bf*)(Wr + (size_t)16 * K + kk);
    v16bf b2 = *(const v16bf*)(Wr + (size_t)32 * K + kk);
    v16bf b3 = *(const v16bf*)(Wr + (size_t)48 * K + kk);
    acc[0][0] = WMMA_BF16(a0, b0, acc[0][0]);
    acc[0][1] = WMMA_BF16(a0, b1, acc[0][1]);
    acc[0][2] = WMMA_BF16(a0, b2, acc[0][2]);
    acc[0][3] = WMMA_BF16(a0, b3, acc[0][3]);
    acc[1][0] = WMMA_BF16(a1, b0, acc[1][0]);
    acc[1][1] = WMMA_BF16(a1, b1, acc[1][1]);
    acc[1][2] = WMMA_BF16(a1, b2, acc[1][2]);
    acc[1][3] = WMMA_BF16(a1, b3, acc[1][3]);
  }

  #pragma unroll
  for (int mt = 0; mt < 2; ++mt)
    #pragma unroll
    for (int nt = 0; nt < 4; ++nt)
      #pragma unroll
      for (int j = 0; j < 8; ++j) {
        int row = m0 + 16 * mt + j + hi8;
        int col = n0 + 16 * nt + lq;
        Cout[(size_t)row * N + col] = acc[mt][nt][j];
      }
}

// ---------------- causal depthwise conv (K=3) + bias, layout fixups -------
// q -> bf16 (B*T, C) ; k -> bf16 (B, T, HD) ; v -> bf16 transposed (B, HD, T)
__global__ void __launch_bounds__(256) qkv_conv(const float* __restrict__ qkv,
                                                const float* __restrict__ qw, const float* __restrict__ qb,
                                                const float* __restrict__ kw, const float* __restrict__ kb,
                                                const float* __restrict__ vw, const float* __restrict__ vb,
                                                __bf16* __restrict__ qh,
                                                __bf16* __restrict__ kh,
                                                __bf16* __restrict__ vT) {
  int idx = blockIdx.x * blockDim.x + threadIdx.x;
  if (idx >= B_ * T_ * O_) return;
  int ch = idx % O_;
  int t  = (idx / O_) % T_;
  int b  = idx / (O_ * T_);
  const float* base = qkv + (size_t)(b * T_ + t) * O_ + ch;
  float x0 = (t >= 2) ? base[-2 * O_] : 0.f;
  float x1 = (t >= 1) ? base[-O_]     : 0.f;
  float x2 = base[0];
  if (ch < C_) {
    float r = qw[ch*3+0]*x0 + qw[ch*3+1]*x1 + qw[ch*3+2]*x2 + qb[ch];
    qh[(size_t)(b * T_ + t) * C_ + ch] = (__bf16)r;
  } else if (ch < C_ + HD_) {
    int d = ch - C_;
    float r = kw[d*3+0]*x0 + kw[d*3+1]*x1 + kw[d*3+2]*x2 + kb[d];
    kh[(size_t)(b * T_ + t) * HD_ + d] = (__bf16)r;
  } else {
    int d = ch - C_ - HD_;
    float r = vw[d*3+0]*x0 + vw[d*3+1]*x1 + vw[d*3+2]*x2 + vb[d];
    vT[(size_t)b * HD_ * T_ + (size_t)d * T_ + t] = (__bf16)r;
  }
}

// ---------------- flash MQA: one wave per (b, h, 16-query tile) -----------
__global__ void __launch_bounds__(256) mqa_attn(const __bf16* __restrict__ qh,
                                                const __bf16* __restrict__ kh,
                                                const __bf16* __restrict__ vT,
                                                __bf16* __restrict__ yh) {
  __shared__ __align__(64) __bf16 Plds[8 * 16 * 32];
  int waveInBlk = threadIdx.x >> 5;
  int lane      = threadIdx.x & 31;
  int wave      = blockIdx.x * 8 + waveInBlk;   // total = B*H*(T/16) = 4096
  int qi = wave & 127;
  int h  = (wave >> 7) & 15;
  int b  = wave >> 11;
  int qb = qi * 16;
  int lq  = lane & 15;
  int hs  = (lane >> 4) << 4;
  int hi8 = (lane >> 4) << 3;
  __bf16* P = Plds + waveInBlk * (16 * 32);

  // Q fragments for the two K=32 steps covering HD=64
  const __bf16* qbase = qh + (size_t)(b * T_ + qb + lq) * C_ + h * HD_ + hs;
  v16bf qa0 = *(const v16bf*)(qbase);
  v16bf qa1 = *(const v16bf*)(qbase + 32);

  const __bf16* kbB = kh + (size_t)b * T_ * HD_;
  const __bf16* vbB = vT + (size_t)b * HD_ * T_;

  v8f acc[4] = {};
  float mrow[8], lrow[8];
  #pragma unroll
  for (int j = 0; j < 8; ++j) { mrow[j] = -3.0e38f; lrow[j] = 0.f; }

  const float sc = 0.125f * 1.44269504088896340736f;  // 1/sqrt(64) * log2(e)
  int nchunks = (qb + 15) / 32 + 1;

  for (int kc = 0; kc < nchunks; ++kc) {
    int n0 = kc * 32;
    // ---- scores: S = Q * K^T for 16 queries x 32 keys
    v8f s0 = {}, s1 = {};
    {
      const __bf16* kp0 = kbB + (size_t)(n0 + lq) * HD_ + hs;
      const __bf16* kp1 = kbB + (size_t)(n0 + 16 + lq) * HD_ + hs;
      v16bf b00 = *(const v16bf*)(kp0);
      v16bf b01 = *(const v16bf*)(kp0 + 32);
      v16bf b10 = *(const v16bf*)(kp1);
      v16bf b11 = *(const v16bf*)(kp1 + 32);
      s0 = WMMA_BF16(qa0, b00, s0);
      s0 = WMMA_BF16(qa1, b01, s0);
      s1 = WMMA_BF16(qa0, b10, s1);
      s1 = WMMA_BF16(qa1, b11, s1);
    }
    bool maskChunk = (kc == nchunks - 1);   // only diagonal chunk intersects
    float newm[8];
    #pragma unroll
    for (int j = 0; j < 8; ++j) {
      float a = s0[j] * sc, c = s1[j] * sc;
      if (maskChunk) {
        int qidx = qb + j + hi8;
        if (n0 + lq > qidx)      a = -3.0e38f;
        if (n0 + 16 + lq > qidx) c = -3.0e38f;
      }
      s0[j] = a; s1[j] = c;
      float mx = fmaxf(a, c);
      mx = fmaxf(mx, __shfl_xor(mx, 1, 32));
      mx = fmaxf(mx, __shfl_xor(mx, 2, 32));
      mx = fmaxf(mx, __shfl_xor(mx, 4, 32));
      mx = fmaxf(mx, __shfl_xor(mx, 8, 32));
      newm[j] = fmaxf(mrow[j], mx);
    }
    // ---- exp + row-sum + rescale; stage P into LDS in A-fragment layout
    #pragma unroll
    for (int j = 0; j < 8; ++j) {
      float p0 = exp2f(s0[j] - newm[j]);
      float p1 = exp2f(s1[j] - newm[j]);
      int m = j + hi8;
      P[m * 32 + lq]      = (__bf16)p0;
      P[m * 32 + 16 + lq] = (__bf16)p1;
      float rs = p0 + p1;
      rs += __shfl_xor(rs, 1, 32);
      rs += __shfl_xor(rs, 2, 32);
      rs += __shfl_xor(rs, 4, 32);
      rs += __shfl_xor(rs, 8, 32);
      float scale = exp2f(mrow[j] - newm[j]);
      lrow[j] = lrow[j] * scale + rs;
      mrow[j] = newm[j];
      #pragma unroll
      for (int dt = 0; dt < 4; ++dt) acc[dt][j] *= scale;
    }
    asm volatile("s_wait_dscnt 0" ::: "memory");
    v16bf pa = *(const v16bf*)(P + lq * 32 + hs);   // A-frag: row=lq, K half=hs
    // ---- O += P * V   (K = 32 keys, 4 N-tiles over HD=64)
    #pragma unroll
    for (int dt = 0; dt < 4; ++dt) {
      const __bf16* vp = vbB + (size_t)(dt * 16 + lq) * T_ + n0 + hs;
      v16bf bv = *(const v16bf*)(vp);
      acc[dt] = WMMA_BF16(pa, bv, acc[dt]);
    }
  }

  // ---- epilogue: normalize and emit bf16 y for the projection GEMM
  #pragma unroll
  for (int j = 0; j < 8; ++j) {
    float inv = 1.0f / lrow[j];
    int m = j + hi8;
    size_t rowoff = (size_t)(b * T_ + qb + m) * C_ + h * HD_;
    #pragma unroll
    for (int dt = 0; dt < 4; ++dt)
      yh[rowoff + dt * 16 + lq] = (__bf16)(acc[dt][j] * inv);
  }
}

// ---------------- workspace layout (bytes) --------------------------------
constexpr size_t OFF_XH    = 0;                                     // 4096*1024 bf16
constexpr size_t OFF_WQKVH = OFF_XH    + (size_t)M_ * C_ * 2;       // 1152*1024 bf16
constexpr size_t OFF_WPROJ = OFF_WQKVH + (size_t)O_ * C_ * 2;       // 1024*1024 bf16
constexpr size_t OFF_QKV   = OFF_WPROJ + (size_t)C_ * C_ * 2;       // 4096*1152 f32
constexpr size_t OFF_QH    = OFF_QKV   + (size_t)M_ * O_ * 4;       // 4096*1024 bf16
constexpr size_t OFF_KH    = OFF_QH    + (size_t)M_ * C_ * 2;       // 4096*64   bf16
constexpr size_t OFF_VT    = OFF_KH    + (size_t)M_ * HD_ * 2;      // 2*64*2048 bf16
constexpr size_t OFF_YH    = OFF_VT    + (size_t)B_ * HD_ * T_ * 2; // 4096*1024 bf16

extern "C" void kernel_launch(void* const* d_in, const int* in_sizes, int n_in,
                              void* d_out, int out_size, void* d_ws, size_t ws_size,
                              hipStream_t stream) {
  const float* x     = (const float*)d_in[0];
  const float* Wqkv  = (const float*)d_in[1];
  const float* qw    = (const float*)d_in[2];
  const float* qb    = (const float*)d_in[3];
  const float* kw    = (const float*)d_in[4];
  const float* kb    = (const float*)d_in[5];
  const float* vw    = (const float*)d_in[6];
  const float* vb    = (const float*)d_in[7];
  const float* Wproj = (const float*)d_in[8];
  float* out = (float*)d_out;

  char* ws = (char*)d_ws;
  __bf16* xh     = (__bf16*)(ws + OFF_XH);
  __bf16* wqkvh  = (__bf16*)(ws + OFF_WQKVH);
  __bf16* wprojh = (__bf16*)(ws + OFF_WPROJ);
  float*  qkvf   = (float*)(ws + OFF_QKV);
  __bf16* qh     = (__bf16*)(ws + OFF_QH);
  __bf16* kh     = (__bf16*)(ws + OFF_KH);
  __bf16* vT     = (__bf16*)(ws + OFF_VT);
  __bf16* yh     = (__bf16*)(ws + OFF_YH);

  // 1) fp32 -> bf16 conversions
  cvt_f32_bf16<<<(M_ * C_ + 255) / 256, 256, 0, stream>>>(x, xh, M_ * C_);
  cvt_f32_bf16<<<(O_ * C_ + 255) / 256, 256, 0, stream>>>(Wqkv, wqkvh, O_ * C_);
  cvt_f32_bf16<<<(C_ * C_ + 255) / 256, 256, 0, stream>>>(Wproj, wprojh, C_ * C_);

  // 2) QKV GEMM: (4096 x 1152 x 1024); 128 m-groups * 18 n-groups = 2304 waves
  gemm_bf16_f32<<<(128 * 18) / 8, 256, 0, stream>>>(xh, wqkvh, qkvf, M_, O_, C_);

  // 3) causal depthwise conv + bias (+ v transpose)
  qkv_conv<<<(B_ * T_ * O_) / 256, 256, 0, stream>>>(qkvf, qw, qb, kw, kb, vw, vb,
                                                     qh, kh, vT);

  // 4) flash MQA: 4096 waves = 512 blocks of 8 waves
  mqa_attn<<<512, 256, 0, stream>>>(qh, kh, vT, yh);

  // 5) output projection: (4096 x 1024 x 1024); 128*16 = 2048 waves
  gemm_bf16_f32<<<(128 * 16) / 8, 256, 0, stream>>>(yh, wprojh, out, M_, C_, C_);
}
